// GFocalV2MultiScaleHead_19825569038782
// MI455X (gfx1250) — compile-verified, matrix operand order checked by reading, not run on hardware
//
#include <hip/hip_runtime.h>
#include <hip/hip_bf16.h>

// ---------------------------------------------------------------------------
// GFocalV2 multi-scale head for MI455X (gfx1250, wave32, WMMA bf16)
//   tokens T = 32*96*96 = 294912, C = HID = 256, reg out = 4*17 = 68
//   d_out layout: [cls_score T][box T*4][reg_pred T*68]
// ---------------------------------------------------------------------------

#define TOKENS   294912
#define C_DIM    256
#define HID_DIM  256
#define NBINS    17
#define O3       68          // 4 * 17
#define O3P      80          // padded to 5 WMMA n-tiles
#define MT       128         // tokens per workgroup
#define THREADS  256         // 8 waves (wave32)

#define XB_ELEMS (MT * C_DIM)        // 32768 bf16 = 64 KB
#define WL_ELEMS (HID_DIM * C_DIM)   // 65536 bf16 = 128 KB

typedef __attribute__((ext_vector_type(16))) __bf16 v16bf;
typedef __attribute__((ext_vector_type(8)))  float  v8f;

// ------------------------- device helpers ----------------------------------

__device__ __forceinline__ void stage_w(__bf16* __restrict__ Wl,
                                        const float* __restrict__ w,
                                        int total, int valid)
{
    // fp32 (global, L2-hot) -> bf16 in LDS; rows past `valid` zero-padded
    for (int i = threadIdx.x * 4; i < total; i += THREADS * 4) {
        float4 v = make_float4(0.f, 0.f, 0.f, 0.f);
        if (i < valid) v = *(const float4*)(w + i);
        __bf16 b4[4] = { (__bf16)v.x, (__bf16)v.y, (__bf16)v.z, (__bf16)v.w };
        *(uint2*)(Wl + i) = *(const uint2*)b4;
    }
}

// One 256->256 layer: each wave computes a 16x256 slab with WMMA, ReLU+bias,
// writes bf16 back to LDS for the next layer.
__device__ __forceinline__ void mlp_layer256(
    const __bf16* __restrict__ A,     // LDS, MT x 256 (row-major, K contig)
    const __bf16* __restrict__ Wl,    // LDS, 256(out) x 256(in)
    const float*  __restrict__ bias,  // global, 256
    __bf16* __restrict__ dst)         // LDS, MT x 256
{
    const int lane  = threadIdx.x & 31;
    const int wave  = threadIdx.x >> 5;
    const int l15   = lane & 15;
    const int khalf = (lane >> 4) * 16;
    const int wrow  = wave * 16;

    v8f acc[16];
    #pragma unroll
    for (int nt = 0; nt < 16; ++nt) acc[nt] = v8f{0,0,0,0,0,0,0,0};

    #pragma unroll
    for (int k0 = 0; k0 < C_DIM; k0 += 32) {
        const v16bf a = *(const v16bf*)(A + (wrow + l15) * C_DIM + k0 + khalf);
        #pragma unroll
        for (int nt = 0; nt < 16; ++nt) {
            const v16bf b = *(const v16bf*)(Wl + (nt * 16 + l15) * C_DIM + k0 + khalf);
            acc[nt] = __builtin_amdgcn_wmma_f32_16x16x32_bf16(
                false, a, false, b, (short)0, acc[nt], false, false);
        }
    }

    const int rbase = wrow + ((lane >> 4) << 3);   // lanes 16-31 hold M+8 rows
    #pragma unroll
    for (int nt = 0; nt < 16; ++nt) {
        const int   col = nt * 16 + l15;
        const float bv  = bias[col];
        #pragma unroll
        for (int r = 0; r < 8; ++r) {
            float v = acc[nt][r] + bv;
            v = fmaxf(v, 0.0f);
            dst[(rbase + r) * C_DIM + col] = (__bf16)v;
        }
    }
}

// Final regression layer: 256 -> 68 (padded to 80), writes f32 to global.
__device__ __forceinline__ void mlp_layer_out68(
    const __bf16* __restrict__ A,     // LDS, MT x 256
    const __bf16* __restrict__ Wl,    // LDS, 80 x 256 (rows >= 68 are zero)
    const float*  __restrict__ bias,  // global, 68
    float* __restrict__ out,          // global, token-major x 68
    long m0g)
{
    const int lane  = threadIdx.x & 31;
    const int wave  = threadIdx.x >> 5;
    const int l15   = lane & 15;
    const int khalf = (lane >> 4) * 16;
    const int wrow  = wave * 16;

    v8f acc[5];
    #pragma unroll
    for (int nt = 0; nt < 5; ++nt) acc[nt] = v8f{0,0,0,0,0,0,0,0};

    #pragma unroll
    for (int k0 = 0; k0 < C_DIM; k0 += 32) {
        const v16bf a = *(const v16bf*)(A + (wrow + l15) * C_DIM + k0 + khalf);
        #pragma unroll
        for (int nt = 0; nt < 5; ++nt) {
            const v16bf b = *(const v16bf*)(Wl + (nt * 16 + l15) * C_DIM + k0 + khalf);
            acc[nt] = __builtin_amdgcn_wmma_f32_16x16x32_bf16(
                false, a, false, b, (short)0, acc[nt], false, false);
        }
    }

    const int rbase = wrow + ((lane >> 4) << 3);
    #pragma unroll
    for (int nt = 0; nt < 5; ++nt) {
        const int col = nt * 16 + l15;
        if (col < O3) {
            const float bv = bias[col];
            #pragma unroll
            for (int r = 0; r < 8; ++r) {
                const long row = m0g + rbase + r;
                out[row * O3 + col] = acc[nt][r] + bv;
            }
        }
    }
}

// ------------------------- fused MLP kernel --------------------------------

__global__ __launch_bounds__(THREADS, 1)
void head_mlp_kernel(const float* __restrict__ x,
                     const float* __restrict__ cw1, const float* __restrict__ cb1,
                     const float* __restrict__ cw2, const float* __restrict__ cb2,
                     const float* __restrict__ cw3, const float* __restrict__ cb3,
                     const float* __restrict__ rw1, const float* __restrict__ rb1,
                     const float* __restrict__ rw2, const float* __restrict__ rb2,
                     const float* __restrict__ rw3, const float* __restrict__ rb3,
                     float* __restrict__ regOut,    // T x 68
                     float* __restrict__ clsLogit)  // T
{
    extern __shared__ __align__(16) char smem_raw[];
    __bf16* Xb = (__bf16*)smem_raw;        // 64 KB  : x tile (bf16)
    __bf16* Ha = Xb + XB_ELEMS;            // 64 KB  : hidden ping
    __bf16* Hb = Ha + XB_ELEMS;            // 64 KB  : hidden pong
    __bf16* Wl = Hb + XB_ELEMS;            // 128 KB : current layer weights
                                           // total 320 KB (full WGP LDS)
    const int  tid = threadIdx.x;
    const long m0g = (long)blockIdx.x * MT;

    // stage activation tile fp32 -> bf16 (x read exactly once overall)
    for (int i = tid * 4; i < MT * C_DIM; i += THREADS * 4) {
        const float4 v = *(const float4*)(x + m0g * C_DIM + i);
        __bf16 b4[4] = { (__bf16)v.x, (__bf16)v.y, (__bf16)v.z, (__bf16)v.w };
        *(uint2*)(Xb + i) = *(const uint2*)b4;
    }
    // prefetch next tile's activations into cache (global_prefetch_b8)
    if ((int)blockIdx.x + 1 < (int)gridDim.x) {
        const float* nx = x + (m0g + MT) * C_DIM;
        for (int i = tid * 64; i < MT * C_DIM; i += THREADS * 64)
            __builtin_prefetch(nx + i, 0, 1);
    }

    // ---- regression branch: 256 -> 256 -> 256 -> 68 ----
    __syncthreads();
    stage_w(Wl, rw1, HID_DIM * C_DIM, HID_DIM * C_DIM);   __syncthreads();
    mlp_layer256(Xb, Wl, rb1, Ha);                        __syncthreads();
    stage_w(Wl, rw2, HID_DIM * HID_DIM, HID_DIM * HID_DIM); __syncthreads();
    mlp_layer256(Ha, Wl, rb2, Hb);                        __syncthreads();
    stage_w(Wl, rw3, O3P * HID_DIM, O3 * HID_DIM);        __syncthreads();
    mlp_layer_out68(Hb, Wl, rb3, regOut, m0g);            __syncthreads();

    // ---- classification branch: 256 -> 256 -> 256 -> 1 ----
    stage_w(Wl, cw1, HID_DIM * C_DIM, HID_DIM * C_DIM);   __syncthreads();
    mlp_layer256(Xb, Wl, cb1, Ha);                        __syncthreads();
    stage_w(Wl, cw2, HID_DIM * HID_DIM, HID_DIM * HID_DIM); __syncthreads();
    mlp_layer256(Ha, Wl, cb2, Hb);                        __syncthreads();

    if (tid < MT) {                       // final 256 -> 1 dot (VALU, tiny)
        float s = cb3[0];
        #pragma unroll 8
        for (int k = 0; k < HID_DIM; ++k)
            s += (float)Hb[tid * HID_DIM + k] * cw3[k];
        clsLogit[m0g + tid] = s;
    }
}

// ------------------------- post-processing kernel --------------------------

__global__ __launch_bounds__(256)
void head_post_kernel(const float* __restrict__ regOut,
                      const float* __restrict__ clsLogit,
                      const float* __restrict__ qw1, const float* __restrict__ qb1,
                      const float* __restrict__ qw2, const float* __restrict__ qb2,
                      float* __restrict__ clsScore, float* __restrict__ box)
{
    const long t = (long)blockIdx.x * blockDim.x + threadIdx.x;
    if (t >= TOKENS) return;
    const float* rp = regOut + t * O3;

    float stat[20], boxv[4];
    #pragma unroll
    for (int s = 0; s < 4; ++s) {
        float p[NBINS];
        float mx = -3.4e38f;
        #pragma unroll
        for (int i = 0; i < NBINS; ++i) { p[i] = rp[s * NBINS + i]; mx = fmaxf(mx, p[i]); }
        float sum = 0.f, integ = 0.f;
        #pragma unroll
        for (int i = 0; i < NBINS; ++i) {
            p[i] = __expf(p[i] - mx);
            sum  += p[i];
            integ += p[i] * (float)i;
        }
        const float inv = 1.f / sum;
        boxv[s] = integ * inv * (1.0f / 16.0f);      // integral / reg_max
        #pragma unroll
        for (int i = 0; i < NBINS; ++i) p[i] *= inv;

        // top-4 + mean (no dynamic indexing -> stays in registers)
        float tsum = 0.f;
        #pragma unroll
        for (int j = 0; j < 4; ++j) {
            float mj = -1.f;
            #pragma unroll
            for (int i = 0; i < NBINS; ++i) mj = fmaxf(mj, p[i]);
            stat[s * 5 + j] = mj;
            tsum += mj;
            bool cleared = false;
            #pragma unroll
            for (int i = 0; i < NBINS; ++i)
                if (!cleared && p[i] == mj) { p[i] = -1.f; cleared = true; }
        }
        stat[s * 5 + 4] = tsum * 0.25f;
    }

    // reg_conf: 20 -> 64 (ReLU) -> 1 (sigmoid)
    float q = qb2[0];
    #pragma unroll 4
    for (int o = 0; o < 64; ++o) {
        float h = qb1[o];
        #pragma unroll
        for (int c = 0; c < 20; ++c) h += qw1[o * 20 + c] * stat[c];
        q += qw2[o] * fmaxf(h, 0.f);
    }
    const float quality = 1.f / (1.f + __expf(-q));
    const float cls     = 1.f / (1.f + __expf(-clsLogit[t]));

    clsScore[t] = cls * quality;
    #pragma unroll
    for (int s = 0; s < 4; ++s) box[t * 4 + s] = boxv[s];
}

// ------------------------- host launcher -----------------------------------

extern "C" void kernel_launch(void* const* d_in, const int* in_sizes, int n_in,
                              void* d_out, int out_size, void* d_ws, size_t ws_size,
                              hipStream_t stream)
{
    (void)in_sizes; (void)n_in; (void)out_size; (void)ws_size;

    const float* x   = (const float*)d_in[0];
    const float* cw1 = (const float*)d_in[1];
    const float* cb1 = (const float*)d_in[2];
    const float* cw2 = (const float*)d_in[3];
    const float* cb2 = (const float*)d_in[4];
    const float* cw3 = (const float*)d_in[5];
    const float* cb3 = (const float*)d_in[6];
    const float* rw1 = (const float*)d_in[7];
    const float* rb1 = (const float*)d_in[8];
    const float* rw2 = (const float*)d_in[9];
    const float* rb2 = (const float*)d_in[10];
    const float* rw3 = (const float*)d_in[11];
    const float* rb3 = (const float*)d_in[12];
    const float* qw1 = (const float*)d_in[13];
    const float* qb1 = (const float*)d_in[14];
    const float* qw2 = (const float*)d_in[15];
    const float* qb2 = (const float*)d_in[16];

    float* clsScore = (float*)d_out;                    // T
    float* box      = clsScore + TOKENS;                // T*4
    float* regOut   = clsScore + (size_t)TOKENS * 5;    // T*68
    float* clsLogit = (float*)d_ws;                     // T (scratch)

    const size_t ldsBytes = (size_t)(3 * XB_ELEMS + WL_ELEMS) * sizeof(__bf16); // 320 KB
    (void)hipFuncSetAttribute(reinterpret_cast<const void*>(head_mlp_kernel),
                              hipFuncAttributeMaxDynamicSharedMemorySize,
                              (int)ldsBytes);

    head_mlp_kernel<<<TOKENS / MT, THREADS, ldsBytes, stream>>>(
        x, cw1, cb1, cw2, cb2, cw3, cb3,
        rw1, rb1, rw2, rb2, rw3, rb3,
        regOut, clsLogit);

    head_post_kernel<<<(TOKENS + 255) / 256, 256, 0, stream>>>(
        regOut, clsLogit, qw1, qb1, qw2, qb2, clsScore, box);
}
